// WD_LSTM_2886218023018
// MI455X (gfx1250) — matrix operand order repaired and checked by reading, not compiled, not run on hardware
//
#include <hip/hip_runtime.h>
#include <hip/hip_bf16.h>

// ---------------------------------------------------------------------------
// MI455X (gfx1250, wave32) implementation of 2-layer LSTM LM forward.
//  - bf16 WMMA (v_wmma_f32_16x16x32_bf16) for all GEMMs, f32 accumulate
//  - Tensor Data Mover (tensor_load_to_lds) stages GEMM tiles into LDS
//  - persistent recurrence kernel: Whh slice held in VGPRs across 512 steps
//  - W_out converted to bf16 (102 MB) so it stays resident in the 192MB L2
// ---------------------------------------------------------------------------

typedef __attribute__((ext_vector_type(16))) __bf16 v16bf;
typedef __attribute__((ext_vector_type(8)))  float  v8f;
typedef __attribute__((ext_vector_type(4)))  unsigned int u32x4;
typedef __attribute__((ext_vector_type(8)))  int  i32x8;
typedef __attribute__((ext_vector_type(4)))  int  i32x4;

union FragBF { v16bf v; uint4 q[2]; };

#define Bq    4
#define Tseq  512
#define Dd    512
#define Hh    1024
#define G4H   4096
#define Vv    50000
#define Vpad  50048
#define Mrows 2048    // B*T
#define NWG   32      // workgroups in persistent recurrence kernel

// -------------------------------------------------- TDM 2D tile load -> LDS
// D# per CDNA5 ISA ch.8: group0 = {count/type/lds_addr/global_addr},
// group1 = {data_size=2B, tensor dims, tile dims, dim0 stride}. 2D tile:
// tile_dim0 = tileK contiguous elements, tile_dim1 = tileN rows.
__device__ __forceinline__ void tdm_load_2d(unsigned lds_addr, const __bf16* gptr,
                                            int rowlen, int nrows,
                                            int tileK, int tileN) {
  unsigned long long ga = (unsigned long long)(uintptr_t)gptr;
  u32x4 g0;
  g0[0] = 1u;                                            // count=1 (valid), user mode
  g0[1] = lds_addr;                                      // LDS byte address
  g0[2] = (unsigned)(ga & 0xFFFFFFFFull);                // global_addr[31:0]
  g0[3] = (unsigned)((ga >> 32) & 0x1FFFFFFull) | (2u << 30);  // [56:32] | type=2
  i32x8 g1;
  g1[0] = (1 << 16);                                     // data_size=1 (2 bytes)
  g1[1] = (rowlen & 0xFFFF) << 16;                       // tensor_dim0[15:0]
  g1[2] = ((rowlen >> 16) & 0xFFFF) | ((nrows & 0xFFFF) << 16);  // dim0 hi | dim1 lo
  g1[3] = ((nrows >> 16) & 0xFFFF) | ((tileK & 0xFFFF) << 16);   // dim1 hi | tile_dim0
  g1[4] = tileN & 0xFFFF;                                // tile_dim1 (tile_dim2=0)
  g1[5] = rowlen;                                        // tensor_dim0_stride[31:0]
  g1[6] = 0;                                             // stride hi / dim1 stride lo
  g1[7] = 0;
  i32x4 z4 = {0, 0, 0, 0};
#if __clang_major__ >= 23
  i32x8 z8 = {0, 0, 0, 0, 0, 0, 0, 0};
  __builtin_amdgcn_tensor_load_to_lds(g0, g1, z4, z4, z8, 0);
#else
  __builtin_amdgcn_tensor_load_to_lds(g0, g1, z4, z4, 0);
#endif
}

// -------------------------------------------------- f32 -> bf16 (zero-pad tail)
__global__ void k_cast_bf16(const float* __restrict__ src, __bf16* __restrict__ dst,
                            long n_src, long n_pad) {
  long i = (long)blockIdx.x * blockDim.x + threadIdx.x;
  long stride = (long)gridDim.x * blockDim.x;
  for (; i < n_pad; i += stride)
    dst[i] = (i < n_src) ? (__bf16)src[i] : (__bf16)0.0f;
}

// -------------------------------------------------- zero u32 region
__global__ void k_zero(unsigned* __restrict__ p, long n) {
  long i = (long)blockIdx.x * blockDim.x + threadIdx.x;
  long stride = (long)gridDim.x * blockDim.x;
  for (; i < n; i += stride) p[i] = 0u;
}

// -------------------------------------------------- embedding gather -> bf16
__global__ void k_gather(const int* __restrict__ idx, const float* __restrict__ X,
                         __bf16* __restrict__ emb) {
  long i = (long)blockIdx.x * blockDim.x + threadIdx.x;
  if (i >= (long)Mrows * Dd) return;
  long r = i >> 9, d = i & 511;
  emb[i] = (__bf16)X[(long)idx[r] * Dd + d];
}

// -------------------------------------------------- WMMA GEMM (TDM-staged)
// C[M,N] = A[M,K](bf16) * W[N,K]^T(bf16) + bias[N], f32 out.
// Block: 256 threads = 8 waves, tile 64(M) x 128(N), K-step 32.
// Wave 0 issues two TDM tile loads per K-step; all waves then consume
// fragments from LDS and issue v_wmma_f32_16x16x32_bf16.
__global__ __launch_bounds__(256)
void k_gemm(const __bf16* __restrict__ A, const __bf16* __restrict__ W,
            const float* __restrict__ bias, float* __restrict__ C,
            int K, int Nreal, int ldc) {
  __shared__ __bf16 smA[64 * 32];
  __shared__ __bf16 smB[128 * 32];
  const int tid  = threadIdx.x;
  const int lane = tid & 31, w = tid >> 5;
  const int m2 = w >> 2, n4 = w & 3;
  const int mbase = blockIdx.y * 64;
  const int nbase = blockIdx.x * 128;
  const int lrow = lane & 15, lhi = lane >> 4;
  const unsigned ldsA = (unsigned)(uintptr_t)&smA[0];
  const unsigned ldsB = (unsigned)(uintptr_t)&smB[0];

  v8f acc[2][2] = {};
  const int prow = tid >> 2;  // prefetch helper row

  for (int k0 = 0; k0 < K; k0 += 32) {
    // ---- TDM: DMA A (64x32) and B (128x32) bf16 tiles into LDS (wave 0)
    if (w == 0) {
      tdm_load_2d(ldsA, A + (size_t)mbase * K + k0, K, 64,  32, 64);
      tdm_load_2d(ldsB, W + (size_t)nbase * K + k0, K, 128, 32, 128);
      __builtin_amdgcn_s_wait_tensorcnt(0);
    } else if (k0 + 32 < K) {  // other waves: cache hints for next tiles
      __builtin_prefetch(A + (size_t)(mbase + prow) * K + k0 + 32, 0, 1);
      __builtin_prefetch(W + (size_t)(nbase + prow) * K + k0 + 32, 0, 1);
      __builtin_prefetch(W + (size_t)(nbase + prow + 64) * K + k0 + 32, 0, 1);
    }
    __syncthreads();

    // ---- A fragments (contiguous k order: 16 contiguous k per lane half, 2x16B per lane)
    FragBF af[2], bfr[2];
#pragma unroll
    for (int am = 0; am < 2; ++am) {
      const __bf16* p = smA + (m2 * 32 + am * 16 + lrow) * 32 + lhi * 16;
      af[am].q[0] = ((const uint4*)p)[0];
      af[am].q[1] = *(const uint4*)(p + 8);
    }
    // ---- B fragments (32 contiguous bytes per lane)
#pragma unroll
    for (int bn = 0; bn < 2; ++bn) {
      const __bf16* p = smB + (n4 * 32 + bn * 16 + lrow) * 32 + lhi * 16;
      bfr[bn].q[0] = ((const uint4*)p)[0];
      bfr[bn].q[1] = ((const uint4*)p)[1];
    }
#pragma unroll
    for (int am = 0; am < 2; ++am)
#pragma unroll
      for (int bn = 0; bn < 2; ++bn)
        acc[am][bn] = __builtin_amdgcn_wmma_f32_16x16x32_bf16(
            false, af[am].v, false, bfr[bn].v, (short)0, acc[am][bn], false, false);
    __syncthreads();
  }

  // ---- epilogue: C layout lane=(col%16, rowhalf), VGPR r -> row (lhi*8+r)
#pragma unroll
  for (int am = 0; am < 2; ++am)
#pragma unroll
    for (int bn = 0; bn < 2; ++bn) {
      int col = nbase + n4 * 32 + bn * 16 + lrow;
      if (col < Nreal) {
        float bv = bias ? bias[col] : 0.0f;
#pragma unroll
        for (int r = 0; r < 8; ++r) {
          int row = mbase + m2 * 32 + am * 16 + lhi * 8 + r;
          C[(size_t)row * ldc + col] = acc[am][bn][r] + bv;
        }
      }
    }
}

// -------------------------------------------------- persistent LSTM recurrence
// 32 WGs x 256 thr. WG owns h-slice j in [blk*32, blk*32+32) across all 4 gates.
// Wave (gate,jhalf) holds its 16x1024 Whh slice in 256 VGPRs for all 512 steps.
// h buffers: padded 16x1024 bf16 (rows 4..15 stay zero), ping-pong per step.
__global__ __launch_bounds__(256)
void k_lstm(const float* __restrict__ xg, const __bf16* __restrict__ Whh,
            __bf16* __restrict__ hseq, __bf16* __restrict__ hping,
            __bf16* __restrict__ hpong, unsigned* __restrict__ bar) {
  __shared__ float gsm[4][4][32];   // [gate][batch][j]
  __shared__ float csm[4][32];      // cell state  [batch][j]
  const int tid = threadIdx.x, lane = tid & 31, w = tid >> 5;
  const int gate = w >> 1, jh = w & 1;
  const int j0 = blockIdx.x * 32;
  const int lrow = lane & 15, lhi = lane >> 4;
  const int nrow = gate * Hh + j0 + jh * 16 + lrow;   // W row (gate column)

  // ---- preload Whh slice: 32 K-fragments -> 256 VGPRs, resident all steps
  FragBF bw[32];
#pragma unroll
  for (int k = 0; k < 32; ++k) {
    const __bf16* p = Whh + (size_t)nrow * Hh + k * 32 + lhi * 16;
    bw[k].q[0] = ((const uint4*)p)[0];
    bw[k].q[1] = ((const uint4*)p)[1];
  }
  if (tid < 128) csm[tid >> 5][tid & 31] = 0.0f;

  unsigned* ctr = bar;
  volatile unsigned* phase = bar + 1;

  for (int t = 0; t < Tseq; ++t) {
    // ---- grid-wide sense barrier (phase monotonic; reset by init kernel)
    __syncthreads();
    if (tid == 0) {
      __threadfence();
      unsigned old = atomicAdd(ctr, 1u);
      if (old == NWG - 1) {
        *ctr = 0; __threadfence();
        atomicAdd((unsigned*)(bar + 1), 1u);
      } else {
        while (*phase < (unsigned)(t + 1)) __builtin_amdgcn_s_sleep(1);
      }
    }
    __syncthreads();

    const __bf16* hb = (t & 1) ? hpong : hping;   // read h_{t}
    __bf16*       hw = (t & 1) ? hping : hpong;   // write h_{t+1}

    // ---- g = h @ Whh^T  for this wave's 16 gate-columns (K=1024)
    v8f acc = {};
#pragma unroll
    for (int k = 0; k < 32; ++k) {
      FragBF af;
      const __bf16* p = hb + (size_t)lrow * Hh + k * 32 + lhi * 16;
      af.q[0] = ((const uint4*)p)[0];
      af.q[1] = *(const uint4*)(p + 8);
      acc = __builtin_amdgcn_wmma_f32_16x16x32_bf16(
          false, af.v, false, bw[k].v, (short)0, acc, false, false);
    }
    if (lane < 16) {                 // rows 0..3 are the real batches
#pragma unroll
      for (int r = 0; r < 4; ++r) gsm[gate][r][jh * 16 + lrow] = acc[r];
    }
    __syncthreads();

    // ---- gate math for 4 batches x 32 j
    if (tid < 128) {
      int b = tid >> 5, j = tid & 31, jg = j0 + j;
      size_t xrow = ((size_t)b * Tseq + t) * (size_t)G4H;
      float gi = gsm[0][b][j] + xg[xrow + 0 * Hh + jg];
      float gf = gsm[1][b][j] + xg[xrow + 1 * Hh + jg];
      float gg = gsm[2][b][j] + xg[xrow + 2 * Hh + jg];
      float go = gsm[3][b][j] + xg[xrow + 3 * Hh + jg];
      gi = 1.0f / (1.0f + __expf(-gi));
      gf = 1.0f / (1.0f + __expf(-gf));
      go = 1.0f / (1.0f + __expf(-go));
      gg = tanhf(gg);
      float c = gf * csm[b][j] + gi * gg;
      csm[b][j] = c;
      float h = go * tanhf(c);
      hw[(size_t)b * Hh + jg] = (__bf16)h;
      hseq[((size_t)b * Tseq + t) * (size_t)Hh + jg] = (__bf16)h;
    }
    __syncthreads();
  }
}

// -------------------------------------------------- log_softmax in place (per row)
__global__ __launch_bounds__(256)
void k_logsoftmax(float* __restrict__ out, int N) {
  float* p = out + (size_t)blockIdx.x * N;
  float m = -3.0e38f, s = 0.0f;
  for (int i = threadIdx.x; i < N; i += 256) {   // online max+sumexp
    float x = p[i];
    float mn = fmaxf(m, x);
    s = s * __expf(m - mn) + __expf(x - mn);
    m = mn;
  }
#pragma unroll
  for (int off = 16; off >= 1; off >>= 1) {      // wave32 reduce
    float m2 = __shfl_xor(m, off, 32);
    float s2 = __shfl_xor(s, off, 32);
    float mn = fmaxf(m, m2);
    s = s * __expf(m - mn) + s2 * __expf(m2 - mn);
    m = mn;
  }
  __shared__ float sm[8], ss[8];
  int w = threadIdx.x >> 5, lane = threadIdx.x & 31;
  if (lane == 0) { sm[w] = m; ss[w] = s; }
  __syncthreads();
  if (threadIdx.x == 0) {
    float M = sm[0], S = ss[0];
    for (int i = 1; i < 8; ++i) {
      float mn = fmaxf(M, sm[i]);
      S = S * __expf(M - mn) + ss[i] * __expf(sm[i] - mn);
      M = mn;
    }
    sm[0] = M; ss[0] = logf(S);
  }
  __syncthreads();
  float M = sm[0], L = ss[0];
  for (int i = threadIdx.x; i < N; i += 256) p[i] = p[i] - M - L;
}

// ---------------------------------------------------------------------------
extern "C" void kernel_launch(void* const* d_in, const int* in_sizes, int n_in,
                              void* d_out, int out_size, void* d_ws, size_t ws_size,
                              hipStream_t stream) {
  (void)in_sizes; (void)n_in; (void)out_size; (void)ws_size;
  const int*   token_idx = (const int*)d_in[0];
  const float* X    = (const float*)d_in[1];
  const float* Wih0 = (const float*)d_in[2];
  const float* Whh0 = (const float*)d_in[3];
  const float* b0   = (const float*)d_in[4];
  const float* Wih1 = (const float*)d_in[5];
  const float* Whh1 = (const float*)d_in[6];
  const float* b1   = (const float*)d_in[7];
  const float* Wout = (const float*)d_in[8];
  const float* bout = (const float*)d_in[9];
  float* logits = (float*)d_out;

  // ---- workspace layout (~176 MB total)
  char* base = (char*)d_ws;
  size_t off = 0;
  auto take = [&](size_t bytes) { char* r = base + off; off += (bytes + 255) & ~(size_t)255; return r; };
  __bf16* pWih0 = (__bf16*)take((size_t)G4H * Dd * 2);
  __bf16* pWhh0 = (__bf16*)take((size_t)G4H * Hh * 2);
  __bf16* pWih1 = (__bf16*)take((size_t)G4H * Hh * 2);
  __bf16* pWhh1 = (__bf16*)take((size_t)G4H * Hh * 2);
  __bf16* pWout = (__bf16*)take((size_t)Vpad * Hh * 2);
  __bf16* pEmb  = (__bf16*)take((size_t)Mrows * Dd * 2);
  float*  pXg   = (float*)take((size_t)Mrows * G4H * 4);
  __bf16* pH0   = (__bf16*)take((size_t)Mrows * Hh * 2);
  __bf16* pH1   = (__bf16*)take((size_t)Mrows * Hh * 2);
  char*   pHbuf = take(16 * Hh * 2 * 2 + 256);          // ping | pong | barrier
  __bf16* hping = (__bf16*)pHbuf;
  __bf16* hpong = (__bf16*)(pHbuf + 16 * Hh * 2);
  unsigned* bar = (unsigned*)(pHbuf + 16 * Hh * 2 * 2);
  const long hbuf_u32 = (16 * Hh * 2 * 2 + 256) / 4;

  auto blocks = [](long n) { long b = (n + 255) / 256; return (int)(b > 524288 ? 524288 : b); };

  // ---- weight casts (W_out zero-padded to 50048 rows)
  { long n = (long)G4H * Dd;  k_cast_bf16<<<blocks(n), 256, 0, stream>>>(Wih0, pWih0, n, n); }
  { long n = (long)G4H * Hh;  k_cast_bf16<<<blocks(n), 256, 0, stream>>>(Whh0, pWhh0, n, n); }
  { long n = (long)G4H * Hh;  k_cast_bf16<<<blocks(n), 256, 0, stream>>>(Wih1, pWih1, n, n); }
  { long n = (long)G4H * Hh;  k_cast_bf16<<<blocks(n), 256, 0, stream>>>(Whh1, pWhh1, n, n); }
  { long ns = (long)Vv * Hh, np = (long)Vpad * Hh;
    k_cast_bf16<<<blocks(np), 256, 0, stream>>>(Wout, pWout, ns, np); }

  // ---- embedding gather
  k_gather<<<blocks((long)Mrows * Dd), 256, 0, stream>>>(token_idx, X, pEmb);

  // ---- layer 0: xg0 = emb @ Wih0^T + b0 ; recurrence
  k_gemm<<<dim3(G4H / 128, Mrows / 64), 256, 0, stream>>>(pEmb, pWih0, b0, pXg, Dd, G4H, G4H);
  k_zero<<<blocks(hbuf_u32), 256, 0, stream>>>((unsigned*)pHbuf, hbuf_u32);
  k_lstm<<<NWG, 256, 0, stream>>>(pXg, pWhh0, pH0, hping, hpong, bar);

  // ---- layer 1
  k_gemm<<<dim3(G4H / 128, Mrows / 64), 256, 0, stream>>>(pH0, pWih1, b1, pXg, Hh, G4H, G4H);
  k_zero<<<blocks(hbuf_u32), 256, 0, stream>>>((unsigned*)pHbuf, hbuf_u32);
  k_lstm<<<NWG, 256, 0, stream>>>(pXg, pWhh1, pH1, hping, hpong, bar);

  // ---- output projection (bias fused) + log_softmax in place
  k_gemm<<<dim3(Vpad / 128, Mrows / 64), 256, 0, stream>>>(pH1, pWout, bout, logits, Hh, Vv, Vv);
  k_logsoftmax<<<Mrows, 256, 0, stream>>>(logits, Vv);
}
